// TransformerBlock_74560632259164
// MI455X (gfx1250) — matrix operand order repaired and checked by reading, not compiled, not run on hardware
//
#include <hip/hip_runtime.h>
#include <hip/hip_bf16.h>

typedef __attribute__((ext_vector_type(16))) __bf16 v16bf;
typedef __attribute__((ext_vector_type(8)))  __bf16 v8bf;
typedef __attribute__((ext_vector_type(8)))  float  v8f;

#define CCH 128   // channel count

// ---------------- WMMA fragment helpers (wave32, 16x16x32 bf16) ----------------

// A fragment from a row-major bf16 LDS tile (16 rows x ld cols).
// ISA 7.12.2 16-bit A 16x32 layout: lanes 0-15 -> K {0..7}u{16..23},
// lanes 16-31 -> K {8..15}u{24..31}. Per lane that is two contiguous
// 16-byte runs -> two ds_load_b128.
__device__ __forceinline__ v16bf load_a_bf16(const __bf16* A, int ld, int kbase, int lane) {
    const int row  = lane & 15;
    const int koff = (lane & 16) ? 8 : 0;
    const __bf16* p = A + row * ld + kbase + koff;
    v8bf lo = *(const v8bf*)p;
    v8bf hi = *(const v8bf*)(p + 16);
    return __builtin_shufflevector(lo, hi, 0,1,2,3,4,5,6,7,8,9,10,11,12,13,14,15);
}

// B fragment from the pre-swizzled bf16 weight buffer: tile = 32 lanes x 16
// contiguous bf16 (32 B per lane) -> one 32-byte vector load (2x b128).
__device__ __forceinline__ v16bf load_b_prep(const __bf16* Wt, int tile, int lane) {
    return *(const v16bf*)(Wt + ((size_t)tile * 32 + lane) * 16);
}

__device__ __forceinline__ v8f splat8(float f) {
    v8f c;
#pragma unroll
    for (int r = 0; r < 8; ++r) c[r] = f;
    return c;
}

#define WMMA_BF16(a, b, c) \
    __builtin_amdgcn_wmma_f32_16x16x32_bf16(false, (a), false, (b), (short)0, (c), false, false)

// ---------------- Kernel P: convert f32 KxN weights to fragment layout ----------------
// dst tile (nt,kt) holds B-frag data: lane = (n&15) + 16*h, h=(kRem>>3)&1, elem i<8 -> K = kbase+8h+i, i>=8 -> K = kbase+16+8h+(i-8)

__global__ __launch_bounds__(256) void prep_weight_kernel(const float* __restrict__ W,
                                                          __bf16* __restrict__ Wt,
                                                          int K, int N) {
    int idx = blockIdx.x * 256 + threadIdx.x;
    if (idx >= K * N) return;
    int k = idx / N, n = idx - k * N;
    int kt = k >> 5, krem = k & 31;
    int nt = n >> 4, nrem = n & 15;
    int lane = nrem + ((krem & 8) << 1);     // +16 iff (krem&8)
    int i = (krem & 7) | ((krem & 16) >> 1);
    int nKT = K >> 5;
    size_t off = (((size_t)(nt * nKT + kt) * 32 + lane) << 4) + i;
    Wt[off] = (__bf16)W[(size_t)k * N + n];
}

// ---------------- Kernel 0: init workspace accumulators ----------------

__global__ __launch_bounds__(256) void init_acc_kernel(float* __restrict__ amax,
                                                       float* __restrict__ den,
                                                       float* __restrict__ num,
                                                       unsigned long long cnt) {
    unsigned long long i = (unsigned long long)blockIdx.x * 256ull + threadIdx.x;
    if (i < cnt) { amax[i] = -3.0e38f; den[i] = 0.f; num[i] = 0.f; }
}

// ---------------- Kernel 1: h = relu(x@W_in+b); alpha_src/alpha_dst/v ----------------

__global__ __launch_bounds__(256) void node_proj_kernel(
    const float* __restrict__ x,
    const __bf16* __restrict__ W_in_t, const float* __restrict__ b_in,
    const __bf16* __restrict__ W_src_t, const __bf16* __restrict__ W_dst_t,
    const __bf16* __restrict__ W_lin_t,
    float* __restrict__ alpha_src, float* __restrict__ alpha_dst, float* __restrict__ vals,
    int n)
{
    __shared__ __align__(16) __bf16 sX[128 * CCH];   // 32 KB
    const int tid = threadIdx.x;
    const int r0  = blockIdx.x * 128;

    for (int idx = tid; idx < 128 * CCH; idx += 256) {
        int row = idx >> 7, col = idx & 127;
        float g = (r0 + row < n) ? x[(size_t)(r0 + row) * CCH + col] : 0.f;
        sX[idx] = (__bf16)g;
    }
    __syncthreads();

    const int wave = tid >> 5, lane = tid & 31;
    const int mrow0 = wave * 16;
    const int lrow  = lane & 15;
    const int mOff  = (lane >> 4) << 3;     // C-layout row offset: 0 or 8

    // h = relu(x @ W_in + b_in)
    v8f acc[8];
#pragma unroll
    for (int nt = 0; nt < 8; ++nt) {
        v8f c = splat8(b_in[nt * 16 + lrow]);
#pragma unroll
        for (int kt = 0; kt < 4; ++kt) {
            v16bf a = load_a_bf16(&sX[mrow0 * CCH], CCH, kt * 32, lane);
            v16bf b = load_b_prep(W_in_t, nt * 4 + kt, lane);
            c = WMMA_BF16(a, b, c);
        }
#pragma unroll
        for (int r = 0; r < 8; ++r) c[r] = fmaxf(c[r], 0.f);
        acc[nt] = c;
    }
    // overwrite this wave's rows with h (all A-frag reads for this phase done)
#pragma unroll
    for (int nt = 0; nt < 8; ++nt) {
        int nn = nt * 16 + lrow;
#pragma unroll
        for (int r = 0; r < 8; ++r)
            sX[(mrow0 + r + mOff) * CCH + nn] = (__bf16)acc[nt][r];
    }

    // three bias-free projections of h
#pragma unroll
    for (int p = 0; p < 3; ++p) {
        const __bf16* Wp = (p == 0) ? W_src_t : (p == 1) ? W_dst_t : W_lin_t;
        float*        Op = (p == 0) ? alpha_src : (p == 1) ? alpha_dst : vals;
#pragma unroll
        for (int nt = 0; nt < 8; ++nt) {
            v8f c = {};
#pragma unroll
            for (int kt = 0; kt < 4; ++kt) {
                v16bf a = load_a_bf16(&sX[mrow0 * CCH], CCH, kt * 32, lane);
                v16bf b = load_b_prep(Wp, nt * 4 + kt, lane);
                c = WMMA_BF16(a, b, c);
            }
            int nn = nt * 16 + lrow;
#pragma unroll
            for (int r = 0; r < 8; ++r) {
                int gr = r0 + mrow0 + r + mOff;
                if (gr < n) Op[(size_t)gr * CCH + nn] = c[r];
            }
        }
    }
}

// ---------------- Kernel 2: edge passes ----------------
// PASS 0: a = attn_nn(alpha_i - alpha_j + pos_nn(dp)); atomic max -> amax[dst]
// PASS 1: recompute a & delta; ae = exp(a - amax); atomic add ae -> den,
//         atomic add ae*(v[src]+delta) -> num.

template <int PASS>
__global__ __launch_bounds__(128) void edge_kernel(
    const float* __restrict__ pos, const long long* __restrict__ ei,
    const float* __restrict__ Wp1, const float* __restrict__ bp1,
    const __bf16* __restrict__ Wp2_t, const float* __restrict__ bp2,
    const __bf16* __restrict__ Wa1_t, const float* __restrict__ ba1,
    const __bf16* __restrict__ Wa2_t, const float* __restrict__ ba2,
    const float* __restrict__ alpha_src, const float* __restrict__ alpha_dst,
    const float* __restrict__ vals,
    float* __restrict__ amax, float* __restrict__ den, float* __restrict__ num,
    int nE, int nEtot)
{
    __shared__ __align__(16) __bf16 sHid[4][16 * 64];    // 2 KB/wave
    __shared__ __align__(16) __bf16 sT[4][16 * CCH];     // 4 KB/wave
    __shared__ __align__(16) __bf16 sD[4][16 * CCH];     // 4 KB/wave (PASS 1)
    __shared__ int sSrc[4][16], sDst[4][16];

    const int tid = threadIdx.x, wave = tid >> 5, lane = tid & 31;
    const int eb = (blockIdx.x * 4 + wave) * 16;

    if (lane < 16) {
        int e = eb + lane;
        int ec = (e < nEtot) ? e : (nEtot - 1);
        int s, d;
        if (ec < nE) { s = (int)ei[ec]; d = (int)ei[nE + ec]; }
        else         { s = d = ec - nE; }               // self loops
        sSrc[wave][lane] = s; sDst[wave][lane] = d;
        __builtin_prefetch(ei + ec, 0, 1);              // global_prefetch_b8
    }
    __syncthreads();

    const int lrow = lane & 15;
    const int mOff = (lane >> 4) << 3;

    // pos_nn layer 1 : hidden[16x64] = relu((pos_i - pos_j) @ Wp1 + bp1)
    for (int i = 0; i < 32; ++i) {
        int idx = i * 32 + lane;            // 0..1023
        int m = idx >> 6, hc = idx & 63;
        int gs = sSrc[wave][m], gd = sDst[wave][m];
        float p0 = pos[gd * 3 + 0] - pos[gs * 3 + 0];
        float p1 = pos[gd * 3 + 1] - pos[gs * 3 + 1];
        float p2 = pos[gd * 3 + 2] - pos[gs * 3 + 2];
        float hd = bp1[hc] + p0 * Wp1[hc] + p1 * Wp1[64 + hc] + p2 * Wp1[128 + hc];
        sHid[wave][m * 64 + hc] = (__bf16)fmaxf(hd, 0.f);
    }
    __syncthreads();

    // delta = hid @ Wp2 + bp2 ; apre = delta + alpha_dst[i] - alpha_src[j]
#pragma unroll
    for (int nt = 0; nt < 8; ++nt) {
        int nn = nt * 16 + lrow;
        v8f c = splat8(bp2[nn]);
#pragma unroll
        for (int kt = 0; kt < 2; ++kt) {
            v16bf a = load_a_bf16(sHid[wave], 64, kt * 32, lane);
            v16bf b = load_b_prep(Wp2_t, nt * 2 + kt, lane);
            c = WMMA_BF16(a, b, c);
        }
#pragma unroll
        for (int r = 0; r < 8; ++r) {
            int m = r + mOff;
            float dlt = c[r];
            if (PASS == 1) sD[wave][m * CCH + nn] = (__bf16)dlt;
            int gi = sDst[wave][m], gj = sSrc[wave][m];
            float apre = dlt + alpha_dst[(size_t)gi * CCH + nn]
                             - alpha_src[(size_t)gj * CCH + nn];
            sT[wave][m * CCH + nn] = (__bf16)apre;
        }
    }
    __syncthreads();

    // hid2 = relu(apre @ Wa1 + ba1)   [16x64]
#pragma unroll
    for (int nt = 0; nt < 4; ++nt) {
        int nn = nt * 16 + lrow;
        v8f c = splat8(ba1[nn]);
#pragma unroll
        for (int kt = 0; kt < 4; ++kt) {
            v16bf a = load_a_bf16(sT[wave], CCH, kt * 32, lane);
            v16bf b = load_b_prep(Wa1_t, nt * 4 + kt, lane);
            c = WMMA_BF16(a, b, c);
        }
#pragma unroll
        for (int r = 0; r < 8; ++r)
            sHid[wave][(r + mOff) * 64 + nn] = (__bf16)fmaxf(c[r], 0.f);
    }
    __syncthreads();

    // a = hid2 @ Wa2 + ba2 ; segment reduce via atomics
#pragma unroll
    for (int nt = 0; nt < 8; ++nt) {
        int nn = nt * 16 + lrow;
        v8f c = splat8(ba2[nn]);
#pragma unroll
        for (int kt = 0; kt < 2; ++kt) {
            v16bf a = load_a_bf16(sHid[wave], 64, kt * 32, lane);
            v16bf b = load_b_prep(Wa2_t, nt * 2 + kt, lane);
            c = WMMA_BF16(a, b, c);
        }
#pragma unroll
        for (int r = 0; r < 8; ++r) {
            int m = r + mOff;
            if (eb + m >= nEtot) continue;          // ragged tail
            int gi = sDst[wave][m];
            size_t off = (size_t)gi * CCH + nn;
            if (PASS == 0) {
                __hip_atomic_fetch_max(&amax[off], c[r],
                                       __ATOMIC_RELAXED, __HIP_MEMORY_SCOPE_AGENT);
            } else {
                int gj = sSrc[wave][m];
                float ae = __expf(c[r] - amax[off]);
                atomicAdd(&den[off], ae);
                float msg = ae * (vals[(size_t)gj * CCH + nn]
                                  + (float)sD[wave][m * CCH + nn]);
                atomicAdd(&num[off], msg);
            }
        }
    }
}

// ---------------- Kernel 3: out = relu((num/den) @ W_out + b_out) ----------------

__global__ __launch_bounds__(256) void node_out_kernel(
    const float* __restrict__ num, const float* __restrict__ den,
    const __bf16* __restrict__ W_out_t, const float* __restrict__ b_out,
    float* __restrict__ out, int n)
{
    __shared__ __align__(16) __bf16 sA[128 * CCH];
    const int tid = threadIdx.x;
    const int r0  = blockIdx.x * 128;

    for (int idx = tid; idx < 128 * CCH; idx += 256) {
        int row = idx >> 7, col = idx & 127;
        float g = 0.f;
        if (r0 + row < n) {
            size_t o = (size_t)(r0 + row) * CCH + col;
            g = num[o] / (den[o] + 1e-16f);
        }
        sA[idx] = (__bf16)g;
    }
    __syncthreads();

    const int wave = tid >> 5, lane = tid & 31;
    const int mrow0 = wave * 16;
    const int lrow  = lane & 15;
    const int mOff  = (lane >> 4) << 3;

#pragma unroll
    for (int nt = 0; nt < 8; ++nt) {
        int nn = nt * 16 + lrow;
        v8f c = splat8(b_out[nn]);
#pragma unroll
        for (int kt = 0; kt < 4; ++kt) {
            v16bf a = load_a_bf16(&sA[mrow0 * CCH], CCH, kt * 32, lane);
            v16bf b = load_b_prep(W_out_t, nt * 4 + kt, lane);
            c = WMMA_BF16(a, b, c);
        }
#pragma unroll
        for (int r = 0; r < 8; ++r) {
            int gr = r0 + mrow0 + r + mOff;
            if (gr < n) out[(size_t)gr * CCH + nn] = fmaxf(c[r], 0.f);
        }
    }
}

// ---------------- Host launch ----------------

extern "C" void kernel_launch(void* const* d_in, const int* in_sizes, int n_in,
                              void* d_out, int out_size, void* d_ws, size_t ws_size,
                              hipStream_t stream) {
    const float*     x     = (const float*)d_in[0];
    const float*     pos   = (const float*)d_in[1];
    const long long* ei    = (const long long*)d_in[2];
    const float*     W_in  = (const float*)d_in[3];
    const float*     b_in  = (const float*)d_in[4];
    const float*     W_lin = (const float*)d_in[5];
    const float*     W_src = (const float*)d_in[6];
    const float*     W_dst = (const float*)d_in[7];
    const float*     Wp1   = (const float*)d_in[8];
    const float*     bp1   = (const float*)d_in[9];
    const float*     Wp2   = (const float*)d_in[10];
    const float*     bp2   = (const float*)d_in[11];
    const float*     Wa1   = (const float*)d_in[12];
    const float*     ba1   = (const float*)d_in[13];
    const float*     Wa2   = (const float*)d_in[14];
    const float*     ba2   = (const float*)d_in[15];
    const float*     W_out = (const float*)d_in[16];
    const float*     b_out = (const float*)d_in[17];

    const int n = in_sizes[0] / CCH;          // 40000
    const int E = in_sizes[2] / 2;            // 640000
    const int Etot = E + n;                   // with self-loops
    const size_t NC = (size_t)n * CCH;

    float* ws        = (float*)d_ws;
    float* alpha_src = ws;
    float* alpha_dst = ws + 1 * NC;
    float* vbuf      = ws + 2 * NC;
    float* amax      = ws + 3 * NC;
    float* den       = ws + 4 * NC;
    float* num       = ws + 5 * NC;

    // bf16 fragment-layout weight buffers (16-byte aligned: 6*NC*4 % 16 == 0)
    __bf16* wb      = (__bf16*)(ws + 6 * NC);
    __bf16* W_in_t  = wb;                 // 128x128 = 16384
    __bf16* W_src_t = wb + 1 * 16384;
    __bf16* W_dst_t = wb + 2 * 16384;
    __bf16* W_lin_t = wb + 3 * 16384;
    __bf16* W_out_t = wb + 4 * 16384;
    __bf16* Wp2_t   = wb + 5 * 16384;     // 64x128 = 8192
    __bf16* Wa1_t   = Wp2_t + 8192;       // 128x64 = 8192
    __bf16* Wa2_t   = Wa1_t + 8192;       // 64x128 = 8192

    prep_weight_kernel<<<64, 256, 0, stream>>>(W_in,  W_in_t,  128, 128);
    prep_weight_kernel<<<64, 256, 0, stream>>>(W_src, W_src_t, 128, 128);
    prep_weight_kernel<<<64, 256, 0, stream>>>(W_dst, W_dst_t, 128, 128);
    prep_weight_kernel<<<64, 256, 0, stream>>>(W_lin, W_lin_t, 128, 128);
    prep_weight_kernel<<<64, 256, 0, stream>>>(W_out, W_out_t, 128, 128);
    prep_weight_kernel<<<32, 256, 0, stream>>>(Wp2,   Wp2_t,    64, 128);
    prep_weight_kernel<<<32, 256, 0, stream>>>(Wa1,   Wa1_t,   128,  64);
    prep_weight_kernel<<<32, 256, 0, stream>>>(Wa2,   Wa2_t,    64, 128);

    init_acc_kernel<<<(unsigned)((NC + 255) / 256), 256, 0, stream>>>(amax, den, num, NC);

    node_proj_kernel<<<(n + 127) / 128, 256, 0, stream>>>(
        x, W_in_t, b_in, W_src_t, W_dst_t, W_lin_t, alpha_src, alpha_dst, vbuf, n);

    const int eblocks = (Etot + 63) / 64;
    edge_kernel<0><<<eblocks, 128, 0, stream>>>(
        pos, ei, Wp1, bp1, Wp2_t, bp2, Wa1_t, ba1, Wa2_t, ba2,
        alpha_src, alpha_dst, vbuf, amax, den, num, E, Etot);
    edge_kernel<1><<<eblocks, 128, 0, stream>>>(
        pos, ei, Wp1, bp1, Wp2_t, bp2, Wa1_t, ba1, Wa2_t, ba2,
        alpha_src, alpha_dst, vbuf, amax, den, num, E, Etot);

    node_out_kernel<<<(n + 127) / 128, 256, 0, stream>>>(
        num, den, W_out_t, b_out, (float*)d_out, n);
}
